// VisualAttentionProcess_27444841021652
// MI455X (gfx1250) — matrix operand order repaired and checked by reading, not compiled, not run on hardware
//
#include <hip/hip_runtime.h>

// ---------------------------------------------------------------------------
// Problem constants: B=16, Sq=4096, C=1280, Ce=768, Skv=77 (pad 96), H=20, d=64.
// ---------------------------------------------------------------------------
#define BDIM    16
#define SQ      4096
#define CDIM    1280
#define CE      768
#define SKV     77
#define KPAD    8
#define HEADS   20
#define DHEAD   64
#define SP      96
#define MTOT    (BDIM*SQ)   // 65536

typedef __attribute__((ext_vector_type(16))) __bf16 v16bf;
typedef __attribute__((ext_vector_type(8)))  __bf16 v8bf;
typedef __attribute__((ext_vector_type(8)))  float  v8f;
typedef __attribute__((ext_vector_type(4)))  float  v4f;
typedef unsigned int u32x4 __attribute__((ext_vector_type(4)));
typedef int          i32x8 __attribute__((ext_vector_type(8)));
typedef int          i32x4 __attribute__((ext_vector_type(4)));

// ---------------------------------------------------------------------------
// WMMA fragment loaders (CDNA5 ISA §7.12.2, wave32):
//  A 16x32: lane<16 -> M=lane, K={0..7,16..23}; lanes 16..31 -> K={8..15,24..31}
//  B 32x16: lane<16 -> N=lane, K={0..7,16..23}; lanes 16..31 -> K={8..15,24..31}
// ---------------------------------------------------------------------------
__device__ __forceinline__ v16bf lda_bf16(const __bf16* p) {
    v8bf lo = *(const v8bf*)(p);
    v8bf hi = *(const v8bf*)(p + 16);
    return __builtin_shufflevector(lo, hi, 0,1,2,3,4,5,6,7,8,9,10,11,12,13,14,15);
}
__device__ __forceinline__ v16bf ldb_bf16(const __bf16* p) {
    v8bf lo = *(const v8bf*)(p);
    v8bf hi = *(const v8bf*)(p + 16);
    return __builtin_shufflevector(lo, hi, 0,1,2,3,4,5,6,7,8,9,10,11,12,13,14,15);
}
__device__ __forceinline__ v16bf lda_f32(const float* p) {
    v4f a0 = *(const v4f*)(p);      v4f a1 = *(const v4f*)(p + 4);
    v4f b0 = *(const v4f*)(p + 16); v4f b1 = *(const v4f*)(p + 20);
    v16bf r;
    r[0]=(__bf16)a0[0]; r[1]=(__bf16)a0[1]; r[2]=(__bf16)a0[2]; r[3]=(__bf16)a0[3];
    r[4]=(__bf16)a1[0]; r[5]=(__bf16)a1[1]; r[6]=(__bf16)a1[2]; r[7]=(__bf16)a1[3];
    r[8]=(__bf16)b0[0]; r[9]=(__bf16)b0[1]; r[10]=(__bf16)b0[2]; r[11]=(__bf16)b0[3];
    r[12]=(__bf16)b1[0]; r[13]=(__bf16)b1[1]; r[14]=(__bf16)b1[2]; r[15]=(__bf16)b1[3];
    return r;
}
__device__ __forceinline__ v8f wmma_bf16(v16bf a, v16bf b, v8f c) {
    return __builtin_amdgcn_wmma_f32_16x16x32_bf16(false, a, false, b, (short)0, c, false, false);
}
__device__ __forceinline__ v8f zero8() { v8f z = {0.f,0.f,0.f,0.f,0.f,0.f,0.f,0.f}; return z; }

// ---------------------------------------------------------------------------
// Tensor Data Mover: issue a 2-D bf16 tile DMA global->LDS.
// D# packing per CDNA5 ISA:
//   group0: [1:0]=count=1, [63:32]=lds_addr, [120:64]=global_addr, [127:126]=2
//   group1: [17:16]=data_size(1 => 2B), [79:48]=tensor_dim0, [111:80]=tensor_dim1,
//           [127:112]=tile_dim0, [143:128]=tile_dim1, [207:160]=dim0_stride
//   groups 2/3 zero (<=2D).  Strides/dims in elements.  TENSORcnt tracks it.
// This toolchain exposes the 6-arg builtin form:
//   (u32x4 g0, i32x8 g1, i32x4, i32x4, i32x8, i32 cpol)
// ---------------------------------------------------------------------------
__device__ __forceinline__ void tdm_load_2d_bf16(unsigned lds_off, const void* gptr,
                                                 unsigned tile_d0, unsigned tile_d1,
                                                 unsigned tensor_d0, unsigned tensor_d1,
                                                 unsigned stride0) {
    unsigned long long ga = (unsigned long long)(uintptr_t)gptr;
    u32x4 g0;
    g0[0] = 1u;                                            // count=1
    g0[1] = lds_off;                                       // lds_addr (bytes)
    g0[2] = (unsigned)(ga & 0xFFFFFFFFu);                  // global_addr[31:0]
    g0[3] = (unsigned)((ga >> 32) & 0x1FFFFFFu) | (2u<<30);// addr[56:32] | type=2
    i32x8 g1;
    g1[0] = (int)(1u << 16);                               // data_size = 2 bytes
    g1[1] = (int)((tensor_d0 & 0xFFFFu) << 16);            // tensor_dim0[15:0]
    g1[2] = (int)((tensor_d0 >> 16) | ((tensor_d1 & 0xFFFFu) << 16));
    g1[3] = (int)((tensor_d1 >> 16) | (tile_d0 << 16));    // | tile_dim0
    g1[4] = (int)(tile_d1 & 0xFFFFu);                      // tile_dim1 (tile_dim2=0)
    g1[5] = (int)stride0;                                  // dim0 stride [31:0]
    g1[6] = 0;
    g1[7] = 0;
    i32x4 z4 = {0,0,0,0};
    i32x8 z8 = {0,0,0,0,0,0,0,0};
    __builtin_amdgcn_tensor_load_to_lds(g0, g1, z4, z4, z8, 0);
}
__device__ __forceinline__ unsigned lds_addr_of(const void* p) {
    return (unsigned)(uintptr_t)p;   // generic LDS pointer: addr[31:0] = LDS offset
}

// ---------------------------------------------------------------------------
// Kernel 1: fp32 -> bf16 weight conversion
// ---------------------------------------------------------------------------
__global__ __launch_bounds__(256) void cvt_f32_bf16(const float* __restrict__ src,
                                                    __bf16* __restrict__ dst, int n) {
    int i = blockIdx.x * 256 + threadIdx.x;
    if (i < n) dst[i] = (__bf16)src[i];
}

// ---------------------------------------------------------------------------
// Kernel 2: K/V projections + erase-branch value edit (tiny).
// ---------------------------------------------------------------------------
__global__ __launch_bounds__(256) void kv_kernel(
    const float* __restrict__ enc, const float* __restrict__ Wk,
    const float* __restrict__ Wv, const float* __restrict__ Bpad,
    const float* __restrict__ u_hat, const float* __restrict__ a_hat,
    __bf16* __restrict__ Kb, __bf16* __restrict__ Vt)
{
    const int b = blockIdx.x / SP;
    const int s = blockIdx.x % SP;
    const int t = threadIdx.x;

    if (s >= SKV) {
        for (int c = t; c < CDIM; c += 256) {
            int h = c / DHEAD, dd = c % DHEAD;
            Kb[(((size_t)(b*HEADS + h))*SP + s)*DHEAD + dd] = (__bf16)0.0f;
            Vt[(((size_t)(b*HEADS + h))*DHEAD + dd)*SP + s] = (__bf16)0.0f;
        }
        return;
    }

    __shared__ float se[CE];
    __shared__ float sv[CDIM];
    __shared__ float red[8*256];
    __shared__ float coeffs[KPAD];

    for (int e = t; e < CE; e += 256) se[e] = enc[((size_t)b*SKV + s)*CE + e];
    __syncthreads();

    float kacc[5], vacc[5];
#pragma unroll
    for (int ci = 0; ci < 5; ci++) { kacc[ci] = 0.f; vacc[ci] = 0.f; }
    for (int e = 0; e < CE; e++) {
        float x = se[e];
#pragma unroll
        for (int ci = 0; ci < 5; ci++) {
            int c = t + ci*256;
            kacc[ci] += x * Wk[(size_t)c*CE + e];
            vacc[ci] += x * Wv[(size_t)c*CE + e];
        }
    }
#pragma unroll
    for (int ci = 0; ci < 5; ci++) {
        int c = t + ci*256;
        Kb[(((size_t)(b*HEADS + c/DHEAD))*SP + s)*DHEAD + (c % DHEAD)] = (__bf16)kacc[ci];
        sv[c] = vacc[ci];
    }
    __syncthreads();

    float part[KPAD];
#pragma unroll
    for (int k = 0; k < KPAD; k++) part[k] = 0.f;
#pragma unroll
    for (int ci = 0; ci < 5; ci++) {
        int c = t + ci*256;
        float vv = sv[c];
        const float* bp = Bpad + ((size_t)s*CDIM + c)*KPAD;
#pragma unroll
        for (int k = 0; k < KPAD; k++) part[k] += vv * bp[k];
    }
#pragma unroll
    for (int k = 0; k < KPAD; k++) red[k*256 + t] = part[k];
    __syncthreads();
    for (int off = 128; off > 0; off >>= 1) {
        if (t < off)
#pragma unroll
            for (int k = 0; k < KPAD; k++) red[k*256 + t] += red[k*256 + t + off];
        __syncthreads();
    }
    if (t < KPAD) coeffs[t] = red[t*256];
    __syncthreads();

    float vfree[5];
    float pt = 0.f, pn = 0.f;
#pragma unroll
    for (int ci = 0; ci < 5; ci++) {
        int c = t + ci*256;
        const float* bp = Bpad + ((size_t)s*CDIM + c)*KPAD;
        float vp = 0.f;
#pragma unroll
        for (int k = 0; k < KPAD; k++) vp += coeffs[k] * bp[k];
        float vf = sv[c] - vp;
        vfree[ci] = vf;
        pt += vf * u_hat[(size_t)s*CDIM + c];
        pn += vf * vf;
    }
    red[t] = pt; red[256 + t] = pn;
    __syncthreads();
    for (int off = 128; off > 0; off >>= 1) {
        if (t < off) { red[t] += red[t + off]; red[256 + t] += red[256 + t + off]; }
        __syncthreads();
    }
    float tS = red[0];
    float denom = sqrtf(red[256]) + 1e-8f;
    bool m = (fabsf(tS) / denom) >= 0.1f;

#pragma unroll
    for (int ci = 0; ci < 5; ci++) {
        int c = t + ci*256;
        float vo = vfree[ci];
        if (m) vo = vo - tS * u_hat[(size_t)s*CDIM + c] + 0.5f * tS * a_hat[(size_t)s*CDIM + c];
        float vnew = (sv[c] - vfree[ci]) + vo;
        Vt[(((size_t)(b*HEADS + c/DHEAD))*DHEAD + (c % DHEAD))*SP + s] = (__bf16)vnew;
    }
}

// ---------------------------------------------------------------------------
// GEMM skeleton (block 128x128, 8 waves = 4(M) x 2(N), K-step 32).
// Weight tile [128 x 32] bf16 is TDM double-buffered into LDS; wave 0 drives
// the DMA pipeline with TENSORcnt split-waits; all waves read B-frags via DS.
// A operand (activations) is streamed straight from global.
// ---------------------------------------------------------------------------
__global__ __launch_bounds__(256) void gemm_q(
    const float*  __restrict__ A,   // [M,Kd] fp32 (hidden)
    const __bf16* __restrict__ W,   // [N,Kd] bf16 (Wq)
    __bf16*       __restrict__ Cm,  // [M,N]  bf16 (Qb)
    int M, int N, int Kd)
{
    const int wave = threadIdx.x >> 5, lane = threadIdx.x & 31;
    const int half = lane >> 4, l16 = lane & 15;
    const int wm = wave & 3, wn = wave >> 2;
    const int row0 = blockIdx.x * 128 + wm * 32;
    const int colblk = blockIdx.y * 128;
    const int col0 = colblk + wn * 64;
    const int nk = Kd / 32;

    __shared__ __align__(16) __bf16 Wlds[2][128*32];   // 16 KB double buffer

    v8f acc[2][4];
#pragma unroll
    for (int mt = 0; mt < 2; mt++)
#pragma unroll
        for (int nt = 0; nt < 4; nt++) acc[mt][nt] = zero8();

    if (wave == 0)  // prologue DMA: tile 0
        tdm_load_2d_bf16(lds_addr_of(&Wlds[0][0]), W + (size_t)colblk*Kd,
                         32, 128, 32, 128, (unsigned)Kd);

    for (int it = 0; it < nk; it++) {
        if (wave == 0) {
            if (it + 1 < nk) {
                tdm_load_2d_bf16(lds_addr_of(&Wlds[(it+1)&1][0]),
                                 W + (size_t)colblk*Kd + (size_t)(it+1)*32,
                                 32, 128, 32, 128, (unsigned)Kd);
                __builtin_amdgcn_s_wait_tensorcnt((short)1);   // tile `it` done
            } else {
                __builtin_amdgcn_s_wait_tensorcnt((short)0);
            }
        }
        __syncthreads();                                        // publish tile `it`

        const __bf16* wt = &Wlds[it&1][0];
        v16bf af[2], bf[4];
#pragma unroll
        for (int mt = 0; mt < 2; mt++) {
            const float* p = A + (size_t)(row0 + mt*16 + l16)*Kd + it*32 + half*8;
            af[mt] = lda_f32(p);
            __builtin_prefetch(p + 32, 0, 0);
        }
#pragma unroll
        for (int nt = 0; nt < 4; nt++)
            bf[nt] = ldb_bf16(wt + (size_t)(wn*64 + nt*16 + l16)*32 + half*8);
#pragma unroll
        for (int mt = 0; mt < 2; mt++)
#pragma unroll
            for (int nt = 0; nt < 4; nt++)
                acc[mt][nt] = wmma_bf16(af[mt], bf[nt], acc[mt][nt]);
        __syncthreads();                 // all reads of buffer `it&1` complete
    }

#pragma unroll
    for (int mt = 0; mt < 2; mt++)
#pragma unroll
        for (int nt = 0; nt < 4; nt++)
#pragma unroll
            for (int j = 0; j < 8; j++) {
                int r = row0 + mt*16 + j + half*8;
                int c = col0 + nt*16 + l16;
                Cm[(size_t)r*N + c] = (__bf16)acc[mt][nt][j];
            }
}

// ---------------------------------------------------------------------------
// Attention: per block (b,h,128 queries). TDM stages K [96x64] and Vt [64x96]
// bf16 tiles into LDS once; each wave computes a 16-row tile:
// scores -> masked fp32 softmax (shfl row-reductions) -> P via LDS transpose
// (C-layout -> A-layout) -> P*V.
// ---------------------------------------------------------------------------
__global__ __launch_bounds__(256) void attn_kernel(
    const __bf16* __restrict__ Qb,   // [MTOT,C]
    const __bf16* __restrict__ Kb,   // [B,H,SP,DHEAD]
    const __bf16* __restrict__ Vt,   // [B,H,DHEAD,SP]
    __bf16*       __restrict__ Ab)   // [MTOT,C]
{
    const float scale = 0.125f;
    const int qblocks = SQ / 128;
    const int qblk = blockIdx.x % qblocks;
    const int bh   = blockIdx.x / qblocks;
    const int b = bh / HEADS, h = bh % HEADS;
    const int wave = threadIdx.x >> 5, lane = threadIdx.x & 31;
    const int half = lane >> 4, l16 = lane & 15;
    const int q0 = qblk*128 + wave*16;
    const size_t rowbase = (size_t)b*SQ + q0;

    __shared__ __align__(16) __bf16 Klds[SP*DHEAD];    // 12 KB
    __shared__ __align__(16) __bf16 Vlds[DHEAD*SP];    // 12 KB
    __shared__ __align__(16) __bf16 Plds[8][16][SP];   // 24 KB

    if (wave == 0) {
        const __bf16* Kp = Kb + (size_t)(b*HEADS + h)*SP*DHEAD;
        const __bf16* Vp = Vt + (size_t)(b*HEADS + h)*DHEAD*SP;
        tdm_load_2d_bf16(lds_addr_of(&Klds[0]), Kp, DHEAD, SP, DHEAD, SP, DHEAD);
        tdm_load_2d_bf16(lds_addr_of(&Vlds[0]), Vp, SP, DHEAD, SP, DHEAD, SP);
        __builtin_amdgcn_s_wait_tensorcnt((short)0);
    }

    // Q A-fragments from global while the DMA is in flight
    v16bf qf[2];
#pragma unroll
    for (int k2 = 0; k2 < 2; k2++) {
        const __bf16* p = Qb + (rowbase + l16)*CDIM + h*DHEAD + k2*32 + half*8;
        qf[k2] = lda_bf16(p);
    }
    __syncthreads();                    // K/V tiles visible in LDS

    // scores: 6 N-tiles x 2 K-steps, B-fragments from LDS
    v8f sc[6];
#pragma unroll
    for (int nt = 0; nt < 6; nt++) {
        sc[nt] = zero8();
#pragma unroll
        for (int k2 = 0; k2 < 2; k2++) {
            const __bf16* p = &Klds[(size_t)(nt*16 + l16)*DHEAD + k2*32 + half*8];
            sc[nt] = wmma_bf16(qf[k2], ldb_bf16(p), sc[nt]);
        }
    }

#pragma unroll
    for (int nt = 0; nt < 6; nt++) {
        bool valid = (nt*16 + l16) < SKV;
#pragma unroll
        for (int j = 0; j < 8; j++) sc[nt][j] = valid ? sc[nt][j]*scale : -1e30f;
    }

#pragma unroll
    for (int j = 0; j < 8; j++) {
        float mx = sc[0][j];
#pragma unroll
        for (int nt = 1; nt < 6; nt++) mx = fmaxf(mx, sc[nt][j]);
        for (int off = 1; off < 16; off <<= 1) mx = fmaxf(mx, __shfl_xor(mx, off));
        float sum = 0.f;
#pragma unroll
        for (int nt = 0; nt < 6; nt++) { float e = __expf(sc[nt][j] - mx); sc[nt][j] = e; sum += e; }
        for (int off = 1; off < 16; off <<= 1) sum += __shfl_xor(sum, off);
        float inv = 1.0f / sum;
        int mrow = j + half*8;
#pragma unroll
        for (int nt = 0; nt < 6; nt++)
            Plds[wave][mrow][nt*16 + l16] = (__bf16)(sc[nt][j] * inv);
    }
    asm volatile("s_wait_dscnt 0" ::: "memory");   // wave-local LDS transpose fence

#pragma unroll
    for (int dt = 0; dt < 4; dt++) {
        v8f acc = zero8();
#pragma unroll
        for (int ks = 0; ks < 3; ks++) {
            const __bf16* ap = &Plds[wave][l16][ks*32 + half*8];
            const __bf16* bp = &Vlds[(size_t)(dt*16 + l16)*SP + ks*32 + half*8];
            acc = wmma_bf16(lda_bf16(ap), ldb_bf16(bp), acc);
        }
#pragma unroll
        for (int j = 0; j < 8; j++) {
            size_t r = rowbase + j + half*8;
            int c = h*DHEAD + dt*16 + l16;
            Ab[r*CDIM + c] = (__bf16)acc[j];
        }
    }
}

// ---------------------------------------------------------------------------
// Final GEMM: out = Ab @ Wo^T + bo (bf16 A, fp32 out), TDM-staged weights,
// nontemporal final stores (result is never re-read).
// ---------------------------------------------------------------------------
__global__ __launch_bounds__(256) void gemm_o(
    const __bf16* __restrict__ A,
    const __bf16* __restrict__ W,
    const float*  __restrict__ bias,
    float*        __restrict__ Cm,
    int M, int N, int Kd)
{
    const int wave = threadIdx.x >> 5, lane = threadIdx.x & 31;
    const int half = lane >> 4, l16 = lane & 15;
    const int wm = wave & 3, wn = wave >> 2;
    const int row0 = blockIdx.x * 128 + wm * 32;
    const int colblk = blockIdx.y * 128;
    const int col0 = colblk + wn * 64;
    const int nk = Kd / 32;

    __shared__ __align__(16) __bf16 Wlds[2][128*32];

    v8f acc[2][4];
#pragma unroll
    for (int mt = 0; mt < 2; mt++)
#pragma unroll
        for (int nt = 0; nt < 4; nt++) acc[mt][nt] = zero8();

    if (wave == 0)
        tdm_load_2d_bf16(lds_addr_of(&Wlds[0][0]), W + (size_t)colblk*Kd,
                         32, 128, 32, 128, (unsigned)Kd);

    for (int it = 0; it < nk; it++) {
        if (wave == 0) {
            if (it + 1 < nk) {
                tdm_load_2d_bf16(lds_addr_of(&Wlds[(it+1)&1][0]),
                                 W + (size_t)colblk*Kd + (size_t)(it+1)*32,
                                 32, 128, 32, 128, (unsigned)Kd);
                __builtin_amdgcn_s_wait_tensorcnt((short)1);
            } else {
                __builtin_amdgcn_s_wait_tensorcnt((short)0);
            }
        }
        __syncthreads();

        const __bf16* wt = &Wlds[it&1][0];
        v16bf af[2], bf[4];
#pragma unroll
        for (int mt = 0; mt < 2; mt++)
            af[mt] = lda_bf16(A + (size_t)(row0 + mt*16 + l16)*Kd + it*32 + half*8);
#pragma unroll
        for (int nt = 0; nt < 4; nt++)
            bf[nt] = ldb_bf16(wt + (size_t)(wn*64 + nt*16 + l16)*32 + half*8);
#pragma unroll
        for (int mt = 0; mt < 2; mt++)
#pragma unroll
            for (int nt = 0; nt < 4; nt++)
                acc[mt][nt] = wmma_bf16(af[mt], bf[nt], acc[mt][nt]);
        __syncthreads();
    }

#pragma unroll
    for (int mt = 0; mt < 2; mt++)
#pragma unroll
        for (int nt = 0; nt < 4; nt++)
#pragma unroll
            for (int j = 0; j < 8; j++) {
                int r = row0 + mt*16 + j + half*8;
                int c = col0 + nt*16 + l16;
                __builtin_nontemporal_store(acc[mt][nt][j] + bias[c], &Cm[(size_t)r*N + c]);
            }
}

// ---------------------------------------------------------------------------
// Launcher
// ---------------------------------------------------------------------------
extern "C" void kernel_launch(void* const* d_in, const int* in_sizes, int n_in,
                              void* d_out, int out_size, void* d_ws, size_t ws_size,
                              hipStream_t stream) {
    const float* hs    = (const float*)d_in[0];
    const float* enc   = (const float*)d_in[1];
    const float* Wq    = (const float*)d_in[2];
    const float* Wk    = (const float*)d_in[3];
    const float* Wv    = (const float*)d_in[4];
    const float* Wo    = (const float*)d_in[5];
    const float* bo    = (const float*)d_in[6];
    const float* Bpad  = (const float*)d_in[7];
    const float* u_hat = (const float*)d_in[8];
    const float* a_hat = (const float*)d_in[9];
    float* out = (float*)d_out;

    char* ws = (char*)d_ws;
    __bf16* Wq_b = (__bf16*)(ws + 0);
    __bf16* Wo_b = (__bf16*)(ws + 3276800);
    __bf16* Kb   = (__bf16*)(ws + 6553600);
    __bf16* Vt   = (__bf16*)(ws + 10485760);
    __bf16* Qb   = (__bf16*)(ws + 14417920);
    __bf16* Ab   = (__bf16*)(ws + 182190080);   // total ~350 MB

    const int wn = CDIM*CDIM;
    cvt_f32_bf16<<<(wn + 255)/256, 256, 0, stream>>>(Wq, Wq_b, wn);
    cvt_f32_bf16<<<(wn + 255)/256, 256, 0, stream>>>(Wo, Wo_b, wn);

    kv_kernel<<<BDIM*SP, 256, 0, stream>>>(enc, Wk, Wv, Bpad, u_hat, a_hat, Kb, Vt);

    gemm_q<<<dim3(MTOT/128, CDIM/128), 256, 0, stream>>>(hs, Wq_b, Qb, MTOT, CDIM, CDIM);

    attn_kernel<<<BDIM*HEADS*(SQ/128), 256, 0, stream>>>(Qb, Kb, Vt, Ab);

    gemm_o<<<dim3(MTOT/128, CDIM/128), 256, 0, stream>>>(Ab, Wo_b, bo, out, MTOT, CDIM, CDIM);
}
